// LearnableUpsamplingLayer_54365696033551
// MI455X (gfx1250) — matrix operand order repaired and checked by reading, compiled, not run on hardware
//
#include <hip/hip_runtime.h>

typedef __attribute__((ext_vector_type(16))) _Float16 v16h;
typedef __attribute__((ext_vector_type(8)))  float    v8f;

#define B_    16
#define T_    1024
#define C_    512
#define TPAD  1032        // padded xc row stride (halves): 1032*2B/4 %64 = 4 -> conflict-free
#define WLEN  2048        // padded filter buffer (halves)
#define WOFF  512         // w index k lives at wpad[k + 512]; A index = s - t + 1023

// ---------------------------------------------------------------------------
// Kernel 1: even frames are a pure copy  out[b, 2t, c] = x[b, t, c]
// Fully coalesced float4 over c (C=512 divisible by 4).
// ---------------------------------------------------------------------------
__global__ __launch_bounds__(256)
void upsample_even_copy(const float* __restrict__ x, float* __restrict__ out) {
    long idx = (long)blockIdx.x * blockDim.x + threadIdx.x;      // over B*T*(C/4)
    const long n4 = (long)B_ * T_ * (C_ / 4);
    if (idx >= n4) return;
    int  c4 = (int)(idx % (C_ / 4));
    long bt = idx / (C_ / 4);
    int  t  = (int)(bt % T_);
    long b  = bt / T_;
    const float4* xi = (const float4*)x;
    float4*       oo = (float4*)out;
    oo[(b * (2 * T_) + 2 * t) * (C_ / 4) + c4] = xi[idx];
}

// ---------------------------------------------------------------------------
// Kernel 2: odd frames via per-channel Toeplitz GEMM on WMMA.
//   scaled[b,t,c] = sum_s xc[b,s] * wpad[s - t + 1023]   (zero-padded filter)
// Grid: (C_, 2). Block: 256 (8 waves). Wave handles 4 t-tiles of 16 rows.
// ---------------------------------------------------------------------------
__global__ __launch_bounds__(256)
void upsample_wmma(const float* __restrict__ x, const float* __restrict__ w,
                   float* __restrict__ out) {
    __shared__ _Float16 wpad[WLEN];          // sigmoid filter, zero-padded both sides
    __shared__ _Float16 xc[B_ * TPAD];       // f16 channel slab, padded rows

    const int c    = blockIdx.x;
    const int tgrp = blockIdx.y;             // 0..1
    const int tid  = threadIdx.x;

    // Fill padded filter: wpad[k + 512] = sigmoid(W[k, c]), zeros elsewhere.
    for (int j = tid; j < WLEN; j += 256) {
        float v = 0.0f;
        int   k = j - WOFF;
        if (k >= 0 && k < T_) {
            float ww = w[(long)k * C_ + c];
            v = 1.0f / (1.0f + __expf(-ww));
        }
        wpad[j] = (_Float16)v;
    }
    // Fill channel slab: xc[b][s] = (f16) x[b, s, c]. Strided 4B reads hit L2
    // (entire x = 32MB is L2-resident on the 192MB L2).
    for (int j = tid; j < B_ * T_; j += 256) {
        int b = j >> 10;
        int s = j & (T_ - 1);
        xc[b * TPAD + s] = (_Float16)x[((long)b * T_ + s) * C_ + c];
    }
    __syncthreads();

    const int lane = tid & 31;
    const int wv   = tid >> 5;               // wave 0..7
    const int row  = lane & 15;              // A: M row; B: N column (= batch b)
    const int hi   = lane >> 4;              // lane-group select

    for (int tt = 0; tt < 4; ++tt) {
        const int t_tile = (tgrp * 4 + tt) * 8 + wv;   // 0..63
        const int t0     = t_tile * 16;
        v8f acc = {};

        for (int s0 = 0; s0 < T_; s0 += 32) {
            // ---- A fragment: Toeplitz tile, A[i,kk] = wpad[s0+kk - t0-i + 1023]
            // 16-bit A 16x32 layout: lanes 0-15: V0..3 -> K0..7, V4..7 -> K16..23
            //                        lanes 16-31: V0..3 -> K8..15, V4..7 -> K24..31
            v16h a;
            const int abase = s0 - t0 - row + 1023;    // in [0, 2046]
            #pragma unroll
            for (int v = 0; v < 8; ++v) {
                const int kb = 2 * (v & 3) + ((v >> 2) << 4) + hi * 8;
                a[2 * v]     = wpad[abase + kb];
                a[2 * v + 1] = wpad[abase + kb + 1];
            }
            // ---- B fragment: B[kk, n] = xc[n][s0 + kk]
            // same K layout as A: lanes 0-15 K0..7,K16..23; lanes 16-31 K8..15,K24..31
            v16h bf;
            const _Float16* xrow = &xc[row * TPAD + s0 + hi * 8];
            #pragma unroll
            for (int v = 0; v < 8; ++v) {
                bf[2 * v]     = xrow[2 * (v & 3) + ((v >> 2) << 4)];
                bf[2 * v + 1] = xrow[2 * (v & 3) + ((v >> 2) << 4) + 1];
            }
            // D = A*B + C  (f32 accumulate)
            acc = __builtin_amdgcn_wmma_f32_16x16x32_f16(
                false, a, false, bf, (short)0, acc, false, false);
        }

        // C/D layout: VGPR r -> M = r + 8*hi, N = row(=b). Odd frame 2t+1.
        const int tb = t0 + hi * 8;
        #pragma unroll
        for (int r = 0; r < 8; ++r) {
            long o = ((long)row * (2 * T_) + (2 * (tb + r) + 1)) * C_ + c;
            out[o] = acc[r];
        }
    }
}

// ---------------------------------------------------------------------------
extern "C" void kernel_launch(void* const* d_in, const int* in_sizes, int n_in,
                              void* d_out, int out_size, void* d_ws, size_t ws_size,
                              hipStream_t stream) {
    (void)in_sizes; (void)n_in; (void)out_size; (void)d_ws; (void)ws_size;
    const float* x = (const float*)d_in[0];              // [16, 1024, 512] f32
    const float* w = (const float*)d_in[1];              // [1024, 512] f32
    float*       o = (float*)d_out;                      // [16, 2048, 512] f32

    const long n4 = (long)B_ * T_ * (C_ / 4);
    upsample_even_copy<<<dim3((unsigned)((n4 + 255) / 256)), dim3(256), 0, stream>>>(x, o);
    upsample_wmma<<<dim3(C_, 2), dim3(256), 0, stream>>>(x, w, o);
}
